// AERIALAgent_60395830116966
// MI455X (gfx1250) — matrix operand order repaired and checked by reading, not compiled, not run on hardware
//
#include <hip/hip_runtime.h>
#include <hip/hip_bf16.h>

#define A_N   8192
#define OBSD  520
#define EMB   64
#define HID   128
#define NACT  6

typedef __attribute__((ext_vector_type(16))) __bf16 v16bf;
typedef __attribute__((ext_vector_type(8)))  float  v8f;
typedef __attribute__((ext_vector_type(4)))  unsigned int v4u;
typedef __attribute__((ext_vector_type(4)))  int v4i;
typedef __attribute__((address_space(1))) v4i gv4i;
typedef __attribute__((address_space(3))) v4i lv4i;

union fragU { v16bf v; v4u q[2]; };
union pk4  { __bf16 h[4]; uint2 u; };

#if __has_builtin(__builtin_amdgcn_global_load_async_to_lds_b128) && \
    __has_builtin(__builtin_amdgcn_s_wait_asynccnt)
#define USE_ASYNC_LDS 1
#else
#define USE_ASYNC_LDS 0
#endif

__device__ __forceinline__ int lane_id() { return (int)(threadIdx.x & 31u); }
__device__ __forceinline__ int wave_id() { return (int)(threadIdx.x >> 5u); }

__device__ __forceinline__ v8f wmma_bf16(v16bf a, v16bf b, v8f c) {
  return __builtin_amdgcn_wmma_f32_16x16x32_bf16(
      false, a, false, b, (short)0, c, false, false);
}

// ---- fragment loaders -----------------------------------------------------
__device__ __forceinline__ v16bf load_a_bf(const __bf16* p, int ld) {
  const int lane = lane_id();
  const int m = lane & 15, g = lane >> 4;
  fragU u;
  u.q[0] = *(const v4u*)(p + m * ld + 8 * g);
  u.q[1] = *(const v4u*)(p + m * ld + 16 + 8 * g);
  return u.v;
}

__device__ __forceinline__ v16bf load_a_f32(const float* p, int ld) {
  const int lane = lane_id();
  const int m = lane & 15, g = lane >> 4;
  const float4 f0 = *(const float4*)(p + m * ld + 8 * g);
  const float4 f1 = *(const float4*)(p + m * ld + 8 * g + 4);
  const float4 f2 = *(const float4*)(p + m * ld + 16 + 8 * g);
  const float4 f3 = *(const float4*)(p + m * ld + 16 + 8 * g + 4);
  v16bf a;
  a[0]=(__bf16)f0.x; a[1]=(__bf16)f0.y; a[2]=(__bf16)f0.z; a[3]=(__bf16)f0.w;
  a[4]=(__bf16)f1.x; a[5]=(__bf16)f1.y; a[6]=(__bf16)f1.z; a[7]=(__bf16)f1.w;
  a[8]=(__bf16)f2.x; a[9]=(__bf16)f2.y; a[10]=(__bf16)f2.z; a[11]=(__bf16)f2.w;
  a[12]=(__bf16)f3.x; a[13]=(__bf16)f3.y; a[14]=(__bf16)f3.z; a[15]=(__bf16)f3.w;
  return a;
}

__device__ __forceinline__ v16bf load_b_nmajor(const __bf16* p, int ld) {
  const int lane = lane_id();
  const int n = lane & 15, g = lane >> 4;
  fragU u;
  u.q[0] = *(const v4u*)(p + n * ld + 8 * g);
  u.q[1] = *(const v4u*)(p + n * ld + 16 + 8 * g);
  return u.v;
}

__device__ __forceinline__ float rowmax16(float v) {
#pragma unroll
  for (int m = 1; m < 16; m <<= 1) v = fmaxf(v, __shfl_xor(v, m, 32));
  return v;
}
__device__ __forceinline__ float rowsum16(float v) {
#pragma unroll
  for (int m = 1; m < 16; m <<= 1) v += __shfl_xor(v, m, 32);
  return v;
}

// stage 128x32 fp32 chunk -> bf16 LDS [128][32]
__device__ __forceinline__ void stage_rows32(__bf16* __restrict__ dst,
                                             const float* __restrict__ src,
                                             int src_ld) {
  const int tid = (int)threadIdx.x;
#pragma unroll
  for (int it = 0; it < 4; ++it) {
    const int i = tid + it * 256;
    const int r = i >> 3, k = (i & 7) * 4;
    const float4 f = *(const float4*)(src + (size_t)r * src_ld + k);
    pk4 w;
    w.h[0] = (__bf16)f.x; w.h[1] = (__bf16)f.y;
    w.h[2] = (__bf16)f.z; w.h[3] = (__bf16)f.w;
    *(uint2*)(dst + r * 32 + k) = w.u;
  }
}

// stage a 16-col slice of W[128 x 128] transposed into LDS [16][128]
__device__ __forceinline__ void stage_slice_t128(__bf16* __restrict__ dst,
                                                 const float* __restrict__ W,
                                                 int col0) {
  const int tid = (int)threadIdx.x;
#pragma unroll
  for (int it = 0; it < 2; ++it) {
    const int i = tid + it * 256;
    const int k = i >> 2, c4 = (i & 3) * 4;
    const float4 f = *(const float4*)(W + (size_t)k * HID + col0 + c4);
    dst[(c4 + 0) * HID + k] = (__bf16)f.x;
    dst[(c4 + 1) * HID + k] = (__bf16)f.y;
    dst[(c4 + 2) * HID + k] = (__bf16)f.z;
    dst[(c4 + 3) * HID + k] = (__bf16)f.w;
  }
}

// ===========================================================================
// Kernel 1: h_obs = relu(relu(obs@W1+b1)@W2+b2)  -> bf16 [A_N, EMB]
// ===========================================================================
__global__ __launch_bounds__(256) void k_obs_mlp(
    const float* __restrict__ obs, const float* __restrict__ W1,
    const float* __restrict__ b1, const float* __restrict__ W2,
    const float* __restrict__ b2, __bf16* __restrict__ hobs) {
  __shared__ __attribute__((aligned(16))) __bf16 sA[128 * 32];
  __shared__ __attribute__((aligned(16))) __bf16 sW1t[64 * 32];   // n-major
  __shared__ __attribute__((aligned(16))) __bf16 sX[128 * 64];
  __shared__ __attribute__((aligned(16))) __bf16 sW2t[64 * 64];   // n-major
  const int tid = (int)threadIdx.x, w = wave_id(), lane = lane_id();
  const int n0 = lane & 15, g = lane >> 4;
  const int rowBase = (int)blockIdx.x * 128;

#pragma unroll
  for (int it = 0; it < 4; ++it) {  // sW2t[n*64 + k] = W2[k][n]
    const int i = tid + it * 256;
    const int k = i >> 4, c4 = (i & 15) * 4;
    const float4 f = *(const float4*)(W2 + (size_t)k * 64 + c4);
    sW2t[(c4 + 0) * 64 + k] = (__bf16)f.x;
    sW2t[(c4 + 1) * 64 + k] = (__bf16)f.y;
    sW2t[(c4 + 2) * 64 + k] = (__bf16)f.z;
    sW2t[(c4 + 3) * 64 + k] = (__bf16)f.w;
  }

  v8f acc[4] = {v8f{}, v8f{}, v8f{}, v8f{}};
#pragma unroll 1
  for (int kc = 0; kc < 16; ++kc) {
    const int k0 = kc * 32;
    __syncthreads();
    stage_rows32(sA, obs + (size_t)rowBase * OBSD + k0, OBSD);
#pragma unroll
    for (int it = 0; it < 2; ++it) {   // W1 chunk transposed
      const int i = tid + it * 256;
      const int k = i >> 4, c4 = (i & 15) * 4;
      const float4 f = *(const float4*)(W1 + (size_t)(k0 + k) * EMB + c4);
      sW1t[(c4 + 0) * 32 + k] = (__bf16)f.x;
      sW1t[(c4 + 1) * 32 + k] = (__bf16)f.y;
      sW1t[(c4 + 2) * 32 + k] = (__bf16)f.z;
      sW1t[(c4 + 3) * 32 + k] = (__bf16)f.w;
    }
    __syncthreads();
    const v16bf a  = load_a_bf(sA + w * 16 * 32, 32);
    const v16bf b0 = load_b_nmajor(sW1t + 0 * 512, 32);
    const v16bf b1f = load_b_nmajor(sW1t + 1 * 512, 32);
    const v16bf b2f = load_b_nmajor(sW1t + 2 * 512, 32);
    const v16bf b3 = load_b_nmajor(sW1t + 3 * 512, 32);
    acc[0] = wmma_bf16(a, b0, acc[0]);
    acc[1] = wmma_bf16(a, b1f, acc[1]);
    acc[2] = wmma_bf16(a, b2f, acc[2]);
    acc[3] = wmma_bf16(a, b3, acc[3]);
  }
  // ---- tail chunk: k = 512..519 valid, rest zero-padded ----
  {
    __syncthreads();
    const v4u z = {};
#pragma unroll
    for (int it = 0; it < 2; ++it) ((v4u*)sA)[tid + it * 256] = z;  // 8KB
    ((v4u*)sW1t)[tid] = z;                                          // 4KB
    __syncthreads();
    {
      const int r = tid >> 1, k = (tid & 1) * 4;  // 128 rows x 8 cols
      const float4 f = *(const float4*)(obs + (size_t)(rowBase + r) * OBSD + 512 + k);
      pk4 wv;
      wv.h[0] = (__bf16)f.x; wv.h[1] = (__bf16)f.y;
      wv.h[2] = (__bf16)f.z; wv.h[3] = (__bf16)f.w;
      *(uint2*)(sA + r * 32 + k) = wv.u;
    }
    if (tid < 128) {  // 8 valid k rows of W1
      const int k = tid >> 4, c4 = (tid & 15) * 4;
      const float4 f = *(const float4*)(W1 + (size_t)(512 + k) * EMB + c4);
      sW1t[(c4 + 0) * 32 + k] = (__bf16)f.x;
      sW1t[(c4 + 1) * 32 + k] = (__bf16)f.y;
      sW1t[(c4 + 2) * 32 + k] = (__bf16)f.z;
      sW1t[(c4 + 3) * 32 + k] = (__bf16)f.w;
    }
    __syncthreads();
    const v16bf a  = load_a_bf(sA + w * 16 * 32, 32);
    const v16bf b0 = load_b_nmajor(sW1t + 0 * 512, 32);
    const v16bf b1f = load_b_nmajor(sW1t + 1 * 512, 32);
    const v16bf b2f = load_b_nmajor(sW1t + 2 * 512, 32);
    const v16bf b3 = load_b_nmajor(sW1t + 3 * 512, 32);
    acc[0] = wmma_bf16(a, b0, acc[0]);
    acc[1] = wmma_bf16(a, b1f, acc[1]);
    acc[2] = wmma_bf16(a, b2f, acc[2]);
    acc[3] = wmma_bf16(a, b3, acc[3]);
  }
  __syncthreads();
#pragma unroll
  for (int nt = 0; nt < 4; ++nt) {
    const int n = nt * 16 + n0;
    const float bn = b1[n];
#pragma unroll
    for (int j = 0; j < 8; ++j) {
      const int m = w * 16 + j + 8 * g;
      sX[m * 64 + n] = (__bf16)fmaxf(acc[nt][j] + bn, 0.0f);
    }
  }
  __syncthreads();
  v8f acc2[4] = {v8f{}, v8f{}, v8f{}, v8f{}};
#pragma unroll
  for (int kc = 0; kc < 2; ++kc) {
    const v16bf a  = load_a_bf(sX + w * 16 * 64 + kc * 32, 64);
    const v16bf b0 = load_b_nmajor(sW2t + 0 * 16 * 64 + kc * 32, 64);
    const v16bf b1f = load_b_nmajor(sW2t + 1 * 16 * 64 + kc * 32, 64);
    const v16bf b2f = load_b_nmajor(sW2t + 2 * 16 * 64 + kc * 32, 64);
    const v16bf b3 = load_b_nmajor(sW2t + 3 * 16 * 64 + kc * 32, 64);
    acc2[0] = wmma_bf16(a, b0, acc2[0]);
    acc2[1] = wmma_bf16(a, b1f, acc2[1]);
    acc2[2] = wmma_bf16(a, b2f, acc2[2]);
    acc2[3] = wmma_bf16(a, b3, acc2[3]);
  }
#pragma unroll
  for (int nt = 0; nt < 4; ++nt) {
    const int n = nt * 16 + n0;
    const float bn = b2[n];
#pragma unroll
    for (int j = 0; j < 8; ++j) {
      const int m = rowBase + w * 16 + j + 8 * g;
      hobs[(size_t)m * EMB + n] = (__bf16)fmaxf(acc2[nt][j] + bn, 0.0f);
    }
  }
}

// ===========================================================================
// Kernel 2: bp = beliefs@Wb + bb  -> bf16 [A_N, EMB]
// ===========================================================================
__global__ __launch_bounds__(256) void k_bp(
    const float* __restrict__ beliefs, const float* __restrict__ Wb,
    const float* __restrict__ bb, __bf16* __restrict__ bp) {
  __shared__ __attribute__((aligned(16))) __bf16 sA[128 * 32];
  __shared__ __attribute__((aligned(16))) __bf16 sWbt[64 * 128];  // n-major
  const int tid = (int)threadIdx.x, w = wave_id(), lane = lane_id();
  const int n0 = lane & 15, g = lane >> 4;
  const int rowBase = (int)blockIdx.x * 128;

#pragma unroll
  for (int it = 0; it < 8; ++it) {  // sWbt[n*128 + k] = Wb[k][n]
    const int i = tid + it * 256;
    const int k = i >> 4, c4 = (i & 15) * 4;
    const float4 f = *(const float4*)(Wb + (size_t)k * EMB + c4);
    sWbt[(c4 + 0) * 128 + k] = (__bf16)f.x;
    sWbt[(c4 + 1) * 128 + k] = (__bf16)f.y;
    sWbt[(c4 + 2) * 128 + k] = (__bf16)f.z;
    sWbt[(c4 + 3) * 128 + k] = (__bf16)f.w;
  }

  v8f acc[4] = {v8f{}, v8f{}, v8f{}, v8f{}};
#pragma unroll 1
  for (int kc = 0; kc < 4; ++kc) {
    __syncthreads();
    stage_rows32(sA, beliefs + (size_t)rowBase * HID + kc * 32, HID);
    __syncthreads();
    const v16bf a  = load_a_bf(sA + w * 16 * 32, 32);
    const v16bf b0 = load_b_nmajor(sWbt + 0 * 16 * 128 + kc * 32, 128);
    const v16bf b1f = load_b_nmajor(sWbt + 1 * 16 * 128 + kc * 32, 128);
    const v16bf b2f = load_b_nmajor(sWbt + 2 * 16 * 128 + kc * 32, 128);
    const v16bf b3 = load_b_nmajor(sWbt + 3 * 16 * 128 + kc * 32, 128);
    acc[0] = wmma_bf16(a, b0, acc[0]);
    acc[1] = wmma_bf16(a, b1f, acc[1]);
    acc[2] = wmma_bf16(a, b2f, acc[2]);
    acc[3] = wmma_bf16(a, b3, acc[3]);
  }
#pragma unroll
  for (int nt = 0; nt < 4; ++nt) {
    const int n = nt * 16 + n0;
    const float bn = bb[n];
#pragma unroll
    for (int j = 0; j < 8; ++j) {
      const int m = rowBase + w * 16 + j + 8 * g;
      bp[(size_t)m * EMB + n] = (__bf16)(acc[nt][j] + bn);
    }
  }
}

// ===========================================================================
// Kernel 3: fused masked self-attention (flash style)
// ===========================================================================
__global__ __launch_bounds__(256) void k_attn(
    const __bf16* __restrict__ bp, __bf16* __restrict__ ctx) {
  __shared__ __attribute__((aligned(16))) __bf16 sQ[128 * 64];
  __shared__ __attribute__((aligned(16))) __bf16 sKV[32 * 64];    // [col][emb]
  __shared__ __attribute__((aligned(16))) __bf16 sKVT[64 * 32];   // [emb][col]
  __shared__ __attribute__((aligned(16))) __bf16 sP[8 * 16 * 32];
  const int tid = (int)threadIdx.x, w = wave_id(), lane = lane_id();
  const int n0 = lane & 15, g = lane >> 4;
  const int rowBase = (int)blockIdx.x * 128;
  __bf16* sPw = sP + w * (16 * 32);

#pragma unroll
  for (int it = 0; it < 4; ++it) {
    const int i = tid + it * 256;
    *(v4u*)(sQ + i * 8) = *(const v4u*)(bp + (size_t)rowBase * 64 + i * 8);
  }
  __syncthreads();
  const v16bf qa0 = load_a_bf(sQ + w * 16 * 64, 64);
  const v16bf qa1 = load_a_bf(sQ + w * 16 * 64 + 32, 64);

  v8f o[4] = {v8f{}, v8f{}, v8f{}, v8f{}};
  float rmax[8], rsum[8];
#pragma unroll
  for (int j = 0; j < 8; ++j) { rmax[j] = -1e30f; rsum[j] = 0.0f; }

#pragma unroll 1
  for (int ct = 0; ct < A_N / 32; ++ct) {
    const int colBase = ct * 32;
    __syncthreads();
#if USE_ASYNC_LDS
    // async DMA the 4KB tile straight into LDS (ASYNCcnt), then transpose copy
    __builtin_amdgcn_global_load_async_to_lds_b128(
        (gv4i*)(bp + (size_t)colBase * 64 + tid * 8),
        (lv4i*)(sKV + tid * 8), 0, 0);
    __builtin_amdgcn_s_wait_asynccnt(0);
    __syncthreads();
#pragma unroll
    for (int it = 0; it < 2; ++it) {
      const int i = tid + it * 256;
      const int c = (i * 4) >> 6, e = (i * 4) & 63;
      pk4 wv;
      wv.u = *(const uint2*)(sKV + c * 64 + e);
      sKVT[(e + 0) * 32 + c] = wv.h[0];
      sKVT[(e + 1) * 32 + c] = wv.h[1];
      sKVT[(e + 2) * 32 + c] = wv.h[2];
      sKVT[(e + 3) * 32 + c] = wv.h[3];
    }
#else
#pragma unroll
    for (int it = 0; it < 2; ++it) {
      const int i = tid + it * 256;
      const int c = (i * 4) >> 6, e = (i * 4) & 63;
      pk4 wv;
      wv.u = *(const uint2*)(bp + (size_t)(colBase + c) * 64 + e);
      *(uint2*)(sKV + c * 64 + e) = wv.u;
      sKVT[(e + 0) * 32 + c] = wv.h[0];
      sKVT[(e + 1) * 32 + c] = wv.h[1];
      sKVT[(e + 2) * 32 + c] = wv.h[2];
      sKVT[(e + 3) * 32 + c] = wv.h[3];
    }
#endif
    if (ct + 1 < A_N / 32)
      __builtin_prefetch(bp + (size_t)(colBase + 32) * 64 + tid * 8, 0, 1);
    __syncthreads();

    // S = Q @ K^T
    const v16bf bk00 = load_b_nmajor(sKV, 64);
    const v16bf bk01 = load_b_nmajor(sKV + 32, 64);
    const v16bf bk10 = load_b_nmajor(sKV + 16 * 64, 64);
    const v16bf bk11 = load_b_nmajor(sKV + 16 * 64 + 32, 64);
    v8f s0 = {}, s1 = {};
    s0 = wmma_bf16(qa0, bk00, s0);
    s0 = wmma_bf16(qa1, bk01, s0);
    s1 = wmma_bf16(qa0, bk10, s1);
    s1 = wmma_bf16(qa1, bk11, s1);

#pragma unroll
    for (int j = 0; j < 8; ++j) {
      float a0 = s0[j] * 0.125f;
      float a1 = s1[j] * 0.125f;
      const int mg = rowBase + w * 16 + j + 8 * g;
      if (mg == colBase + n0)      a0 = -1e30f;
      if (mg == colBase + 16 + n0) a1 = -1e30f;
      const float t  = rowmax16(fmaxf(a0, a1));
      const float nm = fmaxf(rmax[j], t);
      const float sc = __expf(rmax[j] - nm);
      rmax[j] = nm;
      const float p0 = __expf(a0 - nm);
      const float p1 = __expf(a1 - nm);
      rsum[j] = rsum[j] * sc + rowsum16(p0 + p1);
      o[0][j] *= sc; o[1][j] *= sc; o[2][j] *= sc; o[3][j] *= sc;
      const int ml = j + 8 * g;
      sPw[ml * 32 + n0]      = (__bf16)p0;
      sPw[ml * 32 + 16 + n0] = (__bf16)p1;
    }

    // O += P @ V
    const v16bf pa  = load_a_bf(sPw, 32);
    const v16bf bv0 = load_b_nmajor(sKVT + 0 * 16 * 32, 32);
    const v16bf bv1 = load_b_nmajor(sKVT + 1 * 16 * 32, 32);
    const v16bf bv2 = load_b_nmajor(sKVT + 2 * 16 * 32, 32);
    const v16bf bv3 = load_b_nmajor(sKVT + 3 * 16 * 32, 32);
    o[0] = wmma_bf16(pa, bv0, o[0]);
    o[1] = wmma_bf16(pa, bv1, o[1]);
    o[2] = wmma_bf16(pa, bv2, o[2]);
    o[3] = wmma_bf16(pa, bv3, o[3]);
  }

#pragma unroll
  for (int nt = 0; nt < 4; ++nt)
#pragma unroll
    for (int j = 0; j < 8; ++j) {
      const int m = rowBase + w * 16 + j + 8 * g;
      ctx[(size_t)m * EMB + nt * 16 + n0] = (__bf16)(o[nt][j] / rsum[j]);
    }
}

// ===========================================================================
// Kernel 4: fused GRU + output head
// ===========================================================================
__global__ __launch_bounds__(256) void k_gru(
    const __bf16* __restrict__ hobs, const __bf16* __restrict__ ctx,
    const float* __restrict__ beliefs,
    const float* __restrict__ Wg,  const float* __restrict__ bg,
    const float* __restrict__ Wir, const float* __restrict__ Wiz,
    const float* __restrict__ Win, const float* __restrict__ Whr,
    const float* __restrict__ bhr, const float* __restrict__ Whz,
    const float* __restrict__ bhz, const float* __restrict__ Whn,
    const float* __restrict__ bhn, const float* __restrict__ Wout,
    const float* __restrict__ bout,
    float* __restrict__ logits, float* __restrict__ newb) {
  __shared__ __attribute__((aligned(16))) __bf16 sStage[8 * 16 * HID]; // 32KB
  __shared__ __attribute__((aligned(16))) __bf16 sWs[6 * 16 * HID];    // 24KB
  const int tid = (int)threadIdx.x, w = wave_id(), lane = lane_id();
  const int n0 = lane & 15, g = lane >> 4;
  const int rowBase = (int)blockIdx.x * 128;
  __bf16* stg = sStage + w * 16 * HID;

  v16bf xf[4], hf[4];
  xf[0] = load_a_bf(hobs + (size_t)(rowBase + w * 16) * EMB, EMB);
  xf[1] = load_a_bf(hobs + (size_t)(rowBase + w * 16) * EMB + 32, EMB);
  xf[2] = load_a_bf(ctx  + (size_t)(rowBase + w * 16) * EMB, EMB);
  xf[3] = load_a_bf(ctx  + (size_t)(rowBase + w * 16) * EMB + 32, EMB);
#pragma unroll
  for (int kc = 0; kc < 4; ++kc)
    hf[kc] = load_a_f32(beliefs + (size_t)(rowBase + w * 16) * HID + kc * 32, HID);

  // ---- Phase 1: gru_in = [h_obs|ctx] @ Wg + bg -> stage (bf16) ----
#pragma unroll 1
  for (int nt = 0; nt < 8; ++nt) {
    __syncthreads();
    stage_slice_t128(sWs, Wg, nt * 16);
    __syncthreads();
    const v16bf b0 = load_b_nmajor(sWs + 0 * 32, HID);
    const v16bf b1f = load_b_nmajor(sWs + 1 * 32, HID);
    const v16bf b2f = load_b_nmajor(sWs + 2 * 32, HID);
    const v16bf b3 = load_b_nmajor(sWs + 3 * 32, HID);
    v8f acc = {};
    acc = wmma_bf16(xf[0], b0, acc);
    acc = wmma_bf16(xf[1], b1f, acc);
    acc = wmma_bf16(xf[2], b2f, acc);
    acc = wmma_bf16(xf[3], b3, acc);
    const float bn = bg[nt * 16 + n0];
#pragma unroll
    for (int j = 0; j < 8; ++j)
      stg[(j + 8 * g) * HID + nt * 16 + n0] = (__bf16)(acc[j] + bn);
  }
  __syncthreads();
  v16bf gf[4];
#pragma unroll
  for (int kc = 0; kc < 4; ++kc) gf[kc] = load_a_bf(stg + kc * 32, HID);

  // ---- Phase 2: gates per 16-wide output column tile ----
  const float* Wmats[6] = {Wir, Wiz, Win, Whr, Whz, Whn};
#pragma unroll 1
  for (int nt = 0; nt < 8; ++nt) {
    __syncthreads();
#pragma unroll
    for (int sl = 0; sl < 6; ++sl)
      stage_slice_t128(sWs + sl * 2048, Wmats[sl], nt * 16);
    __syncthreads();
    v8f ar = {}, az = {}, an = {}, ahn = {};
#pragma unroll
    for (int kc = 0; kc < 4; ++kc) {
      const v16bf b_ir = load_b_nmajor(sWs + 0 * 2048 + kc * 32, HID);
      const v16bf b_iz = load_b_nmajor(sWs + 1 * 2048 + kc * 32, HID);
      const v16bf b_in = load_b_nmajor(sWs + 2 * 2048 + kc * 32, HID);
      const v16bf b_hr = load_b_nmajor(sWs + 3 * 2048 + kc * 32, HID);
      const v16bf b_hz = load_b_nmajor(sWs + 4 * 2048 + kc * 32, HID);
      const v16bf b_hn = load_b_nmajor(sWs + 5 * 2048 + kc * 32, HID);
      ar  = wmma_bf16(gf[kc], b_ir, ar);
      ar  = wmma_bf16(hf[kc], b_hr, ar);
      az  = wmma_bf16(gf[kc], b_iz, az);
      az  = wmma_bf16(hf[kc], b_hz, az);
      an  = wmma_bf16(gf[kc], b_in, an);
      ahn = wmma_bf16(hf[kc], b_hn, ahn);
    }
    const int col = nt * 16 + n0;
    const float vbhr = bhr[col], vbhz = bhz[col], vbhn = bhn[col];
#pragma unroll
    for (int j = 0; j < 8; ++j) {
      const int row = rowBase + w * 16 + j + 8 * g;
      const float r  = 1.0f / (1.0f + __expf(-(ar[j] + vbhr)));
      const float z  = 1.0f / (1.0f + __expf(-(az[j] + vbhz)));
      const float nn = tanhf(an[j] + r * (ahn[j] + vbhn));
      const float h  = beliefs[(size_t)row * HID + col];
      const float nb = (1.0f - z) * nn + z * h;
      newb[(size_t)row * HID + col] = nb;
      stg[(j + 8 * g) * HID + col] = (__bf16)nb;
    }
  }
  __syncthreads();

  // ---- Phase 3: logits = new_beliefs @ Wout + bout ----
  for (int idx = lane; idx < 16 * NACT; idx += 32) {
    const int rloc = idx / NACT, a = idx % NACT;
    float s = bout[a];
#pragma unroll 4
    for (int k = 0; k < HID; ++k)
      s += (float)stg[rloc * HID + k] * Wout[(size_t)k * NACT + a];
    logits[(size_t)(rowBase + w * 16 + rloc) * NACT + a] = s;
  }
}

// ===========================================================================
extern "C" void kernel_launch(void* const* d_in, const int* in_sizes, int n_in,
                              void* d_out, int out_size, void* d_ws, size_t ws_size,
                              hipStream_t stream) {
  const float* obs     = (const float*)d_in[0];
  const float* beliefs = (const float*)d_in[1];
  const float* W1   = (const float*)d_in[2];
  const float* b1   = (const float*)d_in[3];
  const float* W2   = (const float*)d_in[4];
  const float* b2   = (const float*)d_in[5];
  const float* Wb   = (const float*)d_in[6];
  const float* bb   = (const float*)d_in[7];
  const float* Wg   = (const float*)d_in[8];
  const float* bg   = (const float*)d_in[9];
  const float* Wir  = (const float*)d_in[10];
  const float* Wiz  = (const float*)d_in[11];
  const float* Win  = (const float*)d_in[12];
  const float* Whr  = (const float*)d_in[13];
  const float* bhr  = (const float*)d_in[14];
  const float* Whz  = (const float*)d_in[15];
  const float* bhz  = (const float*)d_in[16];
  const float* Whn  = (const float*)d_in[17];
  const float* bhn  = (const float*)d_in[18];
  const float* Wout = (const float*)d_in[19];
  const float* bout = (const float*)d_in[20];

  __bf16* hobs = (__bf16*)d_ws;
  __bf16* ctx  = hobs + (size_t)A_N * EMB;
  __bf16* bp   = ctx  + (size_t)A_N * EMB;

  float* logits = (float*)d_out;
  float* newb   = logits + (size_t)A_N * NACT;

  dim3 blk(256), grd(A_N / 128);
  k_obs_mlp<<<grd, blk, 0, stream>>>(obs, W1, b1, W2, b2, hobs);
  k_bp<<<grd, blk, 0, stream>>>(beliefs, Wb, bb, bp);
  k_attn<<<grd, blk, 0, stream>>>(bp, ctx);
  k_gru<<<grd, blk, 0, stream>>>(hobs, ctx, beliefs, Wg, bg, Wir, Wiz, Win,
                                 Whr, bhr, Whz, bhz, Whn, bhn, Wout, bout,
                                 logits, newb);
}
